// ZMambaBlock_43370579755491
// MI455X (gfx1250) — matrix operand order repaired and checked by reading, not compiled, not run on hardware
//
#include <hip/hip_runtime.h>

// ---------------- problem constants ----------------
#define B_    2
#define C_    64
#define Z_    32
#define H_    64
#define W_    64
#define HW_   (H_*W_)        // 4096
#define ZHW_  (Z_*HW_)       // 131072
#define NVOX_ (B_*C_*ZHW_)   // 16777216
#define NSEQ_ (B_*HW_)       // 8192 sequences
#define T_    (NSEQ_*Z_)     // 262144 tokens
#define DI_   128            // d_inner
#define DS_   16             // d_state
#define EPS_  1e-5f

typedef __attribute__((ext_vector_type(16))) _Float16 v16h;
typedef __attribute__((ext_vector_type(8)))  _Float16 v8h;
typedef __attribute__((ext_vector_type(8)))  float    v8f;

__device__ __forceinline__ float sigmoidf_(float x){ return 1.f/(1.f+__expf(-x)); }
__device__ __forceinline__ float siluf_(float x){ return x*sigmoidf_(x); }
__device__ __forceinline__ float softplusf_(float x){ return (x>20.f)?x:__logf(1.f+__expf(x)); }

// 16-bit A-matrix 16x32 per-lane K index (ISA 7.12.2): lanes<16 K in {0..7,16..23},
// lanes>=16 K in {8..15,24..31}; 2 halves per VGPR.
__device__ __forceinline__ int kA16_(int lane, int j){
  int hi=(lane>>4)&1; int vg=j>>1; int pos=j&1;
  return (vg<4) ? (vg*2+pos+hi*8) : (16+(vg-4)*2+pos+hi*8);
}
// 16-bit B-matrix 32x16: lane%16 carries N, elements carry K in A's order. Unused helper (returns lane).
__device__ __forceinline__ int kB16_(int lane){ return (lane&15)+((lane>>4)<<4); }

// Load one 16x32 A fragment row from a row-major f16 row pointer (16B aligned).
// K pattern is two contiguous 8-half runs: [hi*8, hi*8+8) and [16+hi*8, 16+hi*8+8).
__device__ __forceinline__ v16h load_a_row_(const _Float16* row, int hi){
  v8h lo = *(const v8h*)(row + hi*8);
  v8h hh = *(const v8h*)(row + 16 + hi*8);
  return __builtin_shufflevector(lo, hh, 0,1,2,3,4,5,6,7,8,9,10,11,12,13,14,15);
}

// Async copy (ASYNCcnt path): global f16 -> LDS, 16B per lane per issue.
// Flat LDS pointers carry the LDS byte offset in the low 32 bits (ISA 10.2).
__device__ __forceinline__ void lds_async_copy_(_Float16* dstLds, const _Float16* src, int halves){
  for (int i = threadIdx.x; i < (halves >> 3); i += blockDim.x){
    unsigned ldsoff = (unsigned)(size_t)(dstLds + i*8);
    unsigned goff   = (unsigned)(i*16);   // bytes
    asm volatile("global_load_async_to_lds_b128 %0, %1, %2"
                 :: "v"(ldsoff), "v"(goff), "s"(src) : "memory");
  }
  asm volatile("s_wait_asynccnt 0" ::: "memory");
}

// ---------------- 0) weight convert + swizzle to B-fragment order ----------------
// out[tile*512 + lane*16 + j] = (f16) W[(nt*16+(lane&15))*K + ks*32 + kA16(lane,j)],
// tile = nt*(K/32)+ks. Each GEMM wave then reads one v16h per fragment.
__global__ void k_cvt_w(const float* __restrict__ w, _Float16* __restrict__ out, int K){
  int idx = blockIdx.x*blockDim.x + threadIdx.x;
  int nk   = K >> 5;
  int tile = idx >> 9;
  int lane = (idx >> 4) & 31;
  int j    = idx & 15;
  int nt = tile / nk, ks = tile % nk;
  int k = ks*32 + kA16_(lane, j);
  int n = nt*16 + (lane&15);
  out[idx] = (_Float16)w[n*K + k];
}

// ---------------- 1) depthwise 1x3x3 conv ----------------
__global__ void k_dwconv(const float* __restrict__ x, const float* __restrict__ dw,
                         float* __restrict__ out){
  int idx = blockIdx.x*blockDim.x + threadIdx.x;
  if (idx >= NVOX_) return;
  int w = idx % W_;
  int h = (idx / W_) % H_;
  int c = (idx / ZHW_) % C_;
  const float* wp = dw + c*9;
  float acc = 0.f;
  #pragma unroll
  for (int dh=-1; dh<=1; ++dh){
    #pragma unroll
    for (int dww=-1; dww<=1; ++dww){
      int hh=h+dh, ww=w+dww;
      if (hh>=0 && hh<H_ && ww>=0 && ww<W_)
        acc += x[idx + dh*W_ + dww] * wp[(dh+1)*3 + (dww+1)];
    }
  }
  out[idx] = acc;
}

// ---------------- 2) pointwise 64x64 via WMMA (weights async-staged in LDS) --------
__global__ void k_pw_wmma(const float* __restrict__ src, const _Float16* __restrict__ wfrag,
                          float* __restrict__ dst){
  __shared__ __align__(32) _Float16 wlds[4096];           // 8 tiles * 512
  lds_async_copy_(wlds, wfrag, 4096);
  __syncthreads();
  int lane = threadIdx.x & 31;
  int gw = blockIdx.x * (blockDim.x>>5) + (threadIdx.x>>5);
  int p0 = gw*16;
  int n   = p0 / ZHW_;
  int pos = p0 % ZHW_;
  v16h a0, a1;                                            // strided channel gather
  #pragma unroll
  for (int j=0;j<16;++j){
    int k = kA16_(lane,j);
    a0[j] = (_Float16)src[(size_t)(n*C_ + k     )*ZHW_ + pos + (lane&15)];
    a1[j] = (_Float16)src[(size_t)(n*C_ + k + 32)*ZHW_ + pos + (lane&15)];
  }
  #pragma unroll
  for (int nt=0; nt<4; ++nt){
    v16h b0 = *(const v16h*)&wlds[(nt*2+0)*512 + lane*16];
    v16h b1 = *(const v16h*)&wlds[(nt*2+1)*512 + lane*16];
    v8f acc = {};
    acc = __builtin_amdgcn_wmma_f32_16x16x32_f16(false,a0,false,b0,(short)0,acc,false,false);
    acc = __builtin_amdgcn_wmma_f32_16x16x32_f16(false,a1,false,b1,(short)0,acc,false,false);
    int o = nt*16 + (lane&15);
    #pragma unroll
    for (int i=0;i<8;++i){
      int mm = i + 8*((lane>>4)&1);
      dst[(size_t)(n*C_ + o)*ZHW_ + pos + mm] = acc[i];
    }
  }
}

// ---------------- instance-norm stats: one block per (n,c) ----------------
__global__ void k_stats(const float* __restrict__ src, float* __restrict__ stats){
  int nc = blockIdx.x;
  const float* p = src + (size_t)nc*ZHW_;
  float s=0.f, s2=0.f;
  for (int i=threadIdx.x; i<ZHW_; i+=blockDim.x){ float v=p[i]; s+=v; s2+=v*v; }
  __shared__ float rs[256], rq[256];
  rs[threadIdx.x]=s; rq[threadIdx.x]=s2; __syncthreads();
  for (int o=128;o>0;o>>=1){
    if ((int)threadIdx.x < o){ rs[threadIdx.x]+=rs[threadIdx.x+o]; rq[threadIdx.x]+=rq[threadIdx.x+o]; }
    __syncthreads();
  }
  if (threadIdx.x==0){
    float mn = rs[0]*(1.f/ZHW_);
    float var = rq[0]*(1.f/ZHW_) - mn*mn;
    stats[nc*2+0] = mn;
    stats[nc*2+1] = rsqrtf(var + EPS_);
  }
}

// ---------------- 3) in1 norm + SiLU + residual add ----------------
__global__ void k_in1_apply(const float* __restrict__ x, const float* __restrict__ y,
                            const float* __restrict__ stats, const float* __restrict__ g,
                            const float* __restrict__ b, float* __restrict__ out){
  int idx = blockIdx.x*blockDim.x + threadIdx.x;
  if (idx>=NVOX_) return;
  int nc = idx / ZHW_;
  int c  = nc % C_;
  float vn = (y[idx]-stats[nc*2])*stats[nc*2+1]*g[c] + b[c];
  out[idx] = x[idx] + siluf_(vn);
}

// ---------------- 4) per-token LayerNorm -> f16 sequence ----------------
__global__ void k_ln(const float* __restrict__ x2, const float* __restrict__ g,
                     const float* __restrict__ b, _Float16* __restrict__ seq){
  int t = blockIdx.x*blockDim.x + threadIdx.x;
  if (t>=T_) return;
  int z  = t & (Z_-1);
  int sq = t >> 5;
  int bb = sq / HW_;
  int hw = sq % HW_;
  size_t base = (size_t)bb*C_*ZHW_ + (size_t)z*HW_ + hw;
  float v[C_];
  float mn=0.f;
  #pragma unroll
  for (int c=0;c<C_;++c){ v[c]=x2[base + (size_t)c*ZHW_]; mn+=v[c]; }
  mn *= (1.f/C_);
  float var=0.f;
  #pragma unroll
  for (int c=0;c<C_;++c){ float d=v[c]-mn; var+=d*d; }
  float r = rsqrtf(var*(1.f/C_) + EPS_);
  #pragma unroll
  for (int c=0;c<C_;++c) seq[(size_t)t*C_ + c] = (_Float16)((v[c]-mn)*r*g[c]+b[c]);
}

// ---------------- 5) in_proj GEMM (T x 64) @ (256 x 64)^T via WMMA ----------------
__global__ void k_inproj_wmma(const _Float16* __restrict__ seq, const _Float16* __restrict__ wfrag,
                              _Float16* __restrict__ xz){
  __shared__ __align__(32) _Float16 wlds[16384];          // 32 tiles * 512 (32 KB)
  lds_async_copy_(wlds, wfrag, 16384);
  __syncthreads();
  int lane = threadIdx.x & 31;
  int hi = (lane>>4)&1;
  int task = blockIdx.x*(blockDim.x>>5) + (threadIdx.x>>5);
  int tile = task >> 2;   // token tile
  int ng   = task & 3;    // 64-output group
  int t0 = tile*16;
  const _Float16* row = seq + (size_t)(t0+(lane&15))*C_;
  v16h a0 = load_a_row_(row, hi);
  v16h a1 = load_a_row_(row + 32, hi);
  #pragma unroll
  for (int nt=0;nt<4;++nt){
    int NT = ng*4 + nt;
    v16h b0 = *(const v16h*)&wlds[(NT*2+0)*512 + lane*16];
    v16h b1 = *(const v16h*)&wlds[(NT*2+1)*512 + lane*16];
    v8f acc={};
    acc=__builtin_amdgcn_wmma_f32_16x16x32_f16(false,a0,false,b0,(short)0,acc,false,false);
    acc=__builtin_amdgcn_wmma_f32_16x16x32_f16(false,a1,false,b1,(short)0,acc,false,false);
    int o = NT*16 + (lane&15);
    #pragma unroll
    for (int i=0;i<8;++i){
      int mm = i + 8*hi;
      xz[(size_t)(t0+mm)*256 + o] = (_Float16)acc[i];
    }
  }
}

// ---------------- 6) conv1d + x_proj + dt + selective scan (1 block / sequence) ----------------
__global__ void __launch_bounds__(128) k_scan(const _Float16* __restrict__ xz,
        const float* __restrict__ cw, const float* __restrict__ cb,
        const float* __restrict__ xpw, const float* __restrict__ dtw,
        const float* __restrict__ dtb, const float* __restrict__ alog,
        const float* __restrict__ dskip, _Float16* __restrict__ yout){
  int s = blockIdx.x;
  int d = threadIdx.x;             // 0..127 = d_inner channel
  __shared__ float xc[Z_][DI_];    // 16 KB
  __shared__ float xd[Z_][40];     // dt(4) | B(16) | C(16), padded
  size_t tb = (size_t)s*Z_;
  float w0=cw[d*4+0], w1=cw[d*4+1], w2=cw[d*4+2], w3=cw[d*4+3];
  float bia=cb[d];
  float xm3=0.f, xm2=0.f, xm1=0.f;
  for (int l=0;l<Z_;++l){          // causal 4-tap conv, rolling registers
    float xm0 = (float)xz[(tb+l)*256 + d];
    float v = bia + w0*xm3 + w1*xm2 + w2*xm1 + w3*xm0;
    xc[l][d] = siluf_(v);
    xm3=xm2; xm2=xm1; xm1=xm0;
  }
  __syncthreads();
  for (int task=d; task<Z_*36; task+=DI_){   // x_proj: 36 outs x 32 tokens
    int l = task/36, e = task%36;
    const float* wr = xpw + e*DI_;
    float acc=0.f;
    #pragma unroll 8
    for (int k=0;k<DI_;++k) acc += wr[k]*xc[l][k];
    xd[l][e]=acc;
  }
  __syncthreads();
  float Ar[DS_];
  #pragma unroll
  for (int q=0;q<DS_;++q) Ar[q] = -__expf(alog[d*DS_+q]);
  float dw0=dtw[d*4+0], dw1=dtw[d*4+1], dw2=dtw[d*4+2], dw3=dtw[d*4+3];
  float db=dtb[d], dk=dskip[d];
  float h[DS_];
  #pragma unroll
  for (int q=0;q<DS_;++q) h[q]=0.f;
  for (int l=0;l<Z_;++l){
    float dlt = softplusf_(dw0*xd[l][0]+dw1*xd[l][1]+dw2*xd[l][2]+dw3*xd[l][3]+db);
    float u  = xc[l][d];
    float du = dlt*u;
    float y  = 0.f;
    #pragma unroll
    for (int q=0;q<DS_;++q){
      float dA = __expf(dlt*Ar[q]);
      h[q] = dA*h[q] + du*xd[l][4+q];
      y += h[q]*xd[l][20+q];
    }
    y += u*dk;
    float zv = (float)xz[(tb+l)*256 + 128 + d];
    y *= zv*sigmoidf_(zv);
    yout[(tb+l)*DI_ + d] = (_Float16)y;
  }
}

// ---------------- 7) fused out_proj (64x128) then proj (64x64) via WMMA + LDS ----------------
__global__ void k_out_wmma(const _Float16* __restrict__ y, const _Float16* __restrict__ wofrag,
                           const _Float16* __restrict__ wpfrag, const float* __restrict__ pb,
                           float* __restrict__ post){
  __shared__ __align__(32) _Float16 wo_lds[8192];        // 16 tiles (16 KB)
  __shared__ __align__(32) _Float16 wp_lds[4096];        // 8 tiles  (8 KB)
  __shared__ __align__(32) _Float16 tmp[8][16][C_+8];    // 18 KB
  lds_async_copy_(wo_lds, wofrag, 8192);
  lds_async_copy_(wp_lds, wpfrag, 4096);
  __syncthreads();
  int lane = threadIdx.x & 31;
  int hi = (lane>>4)&1;
  int wv   = threadIdx.x >> 5;
  int tile = blockIdx.x*(blockDim.x>>5) + wv;
  int t0 = tile*16;
  // GEMM1: y(16x128) @ wo(64x128)^T -> tmp (f16 in LDS)
  const _Float16* row = y + (size_t)(t0+(lane&15))*DI_;
  v16h a[4];
  #pragma unroll
  for (int ks=0;ks<4;++ks) a[ks] = load_a_row_(row + ks*32, hi);
  #pragma unroll
  for (int nt=0;nt<4;++nt){
    v8f acc={};
    #pragma unroll
    for (int ks=0;ks<4;++ks){
      v16h b = *(const v16h*)&wo_lds[(nt*4+ks)*512 + lane*16];
      acc=__builtin_amdgcn_wmma_f32_16x16x32_f16(false,a[ks],false,b,(short)0,acc,false,false);
    }
    int o=nt*16+(lane&15);
    #pragma unroll
    for (int i=0;i<8;++i) tmp[wv][i+8*hi][o]=(_Float16)acc[i];
  }
  __syncthreads();
  // GEMM2: tmp(16x64) @ wp(64x64)^T + pb -> scatter to (b,c,z,h,w)
  const _Float16* trow = &tmp[wv][lane&15][0];
  v16h a0 = load_a_row_(trow, hi);
  v16h a1 = load_a_row_(trow + 32, hi);
  #pragma unroll
  for (int nt=0;nt<4;++nt){
    v16h b0 = *(const v16h*)&wp_lds[(nt*2+0)*512 + lane*16];
    v16h b1 = *(const v16h*)&wp_lds[(nt*2+1)*512 + lane*16];
    v8f acc={};
    acc=__builtin_amdgcn_wmma_f32_16x16x32_f16(false,a0,false,b0,(short)0,acc,false,false);
    acc=__builtin_amdgcn_wmma_f32_16x16x32_f16(false,a1,false,b1,(short)0,acc,false,false);
    int o=nt*16+(lane&15);
    #pragma unroll
    for (int i=0;i<8;++i){
      int mm=i+8*hi;
      int t=t0+mm;
      int z=t&(Z_-1); int sq=t>>5; int bb=sq/HW_; int hw=sq%HW_;
      post[(size_t)(bb*C_+o)*ZHW_ + (size_t)z*HW_ + hw] = acc[i] + pb[o];
    }
  }
}

// ---------------- 8) post instance-norm + residual ----------------
__global__ void k_post(const float* __restrict__ x, const float* __restrict__ post,
                       const float* __restrict__ stats, const float* __restrict__ g,
                       const float* __restrict__ b, float* __restrict__ out){
  int idx = blockIdx.x*blockDim.x + threadIdx.x;
  if (idx>=NVOX_) return;
  int nc = idx/ZHW_; int c = nc%C_;
  out[idx] = x[idx] + (post[idx]-stats[nc*2])*stats[nc*2+1]*g[c] + b[c];
}

// ---------------- host launcher ----------------
extern "C" void kernel_launch(void* const* d_in, const int* in_sizes, int n_in,
                              void* d_out, int out_size, void* d_ws, size_t ws_size,
                              hipStream_t stream){
  const float* x     = (const float*)d_in[0];
  const float* dw_w  = (const float*)d_in[1];
  const float* pw_w  = (const float*)d_in[2];
  const float* in1_g = (const float*)d_in[3];
  const float* in1_b = (const float*)d_in[4];
  const float* ln_g  = (const float*)d_in[5];
  const float* ln_b  = (const float*)d_in[6];
  const float* inpw  = (const float*)d_in[7];
  const float* c1w   = (const float*)d_in[8];
  const float* c1b   = (const float*)d_in[9];
  const float* xpw   = (const float*)d_in[10];
  const float* dtw   = (const float*)d_in[11];
  const float* dtb   = (const float*)d_in[12];
  const float* alog  = (const float*)d_in[13];
  const float* dskip = (const float*)d_in[14];
  const float* outw  = (const float*)d_in[15];
  const float* projw = (const float*)d_in[16];
  const float* projb = (const float*)d_in[17];
  const float* postg = (const float*)d_in[18];
  const float* postb = (const float*)d_in[19];
  float* out = (float*)d_out;
  char* ws = (char*)d_ws;
  const size_t MB = 1024ull*1024ull;
  // workspace layout (with reuse): 320 MB activations + stats + 64 KB f16 weights
  float*    t_dw   = (float*)(ws + 0);        // 64 MB  (freed after pw)
  float*    t_pw   = (float*)(ws + 64*MB);    // 64 MB  (freed after in1_apply)
  float*    t_x2   = (float*)(ws + 128*MB);   // 64 MB  (freed after ln)
  _Float16* t_seq  = (_Float16*)(ws + 0);     // 32 MB  (aliases t_dw)
  _Float16* t_xz   = (_Float16*)(ws + 192*MB);// 128 MB
  _Float16* t_y    = (_Float16*)(ws + 64*MB); // 64 MB  (aliases t_pw)
  float*    t_post = (float*)(ws + 128*MB);   // 64 MB  (aliases t_x2)
  float*    st1    = (float*)(ws + 320*MB);   // 2*B*C floats
  float*    st2    = st1 + 256;
  _Float16* whp    = (_Float16*)(ws + 320*MB + 4096);          // pw      4096 halves
  _Float16* whi    = whp + 4096;                               // in_proj 16384 halves
  _Float16* who    = whi + 16384;                              // out_proj 8192 halves
  _Float16* whj    = who + 8192;                               // proj    4096 halves

  // weight convert + swizzle into WMMA B-fragment order (f16)
  k_cvt_w<<<4096/256, 256, 0, stream>>>(pw_w,   whp, 64);
  k_cvt_w<<<16384/256,256, 0, stream>>>(inpw,   whi, 64);
  k_cvt_w<<<8192/256, 256, 0, stream>>>(outw,   who, 128);
  k_cvt_w<<<4096/256, 256, 0, stream>>>(projw,  whj, 64);

  k_dwconv    <<<NVOX_/256, 256, 0, stream>>>(x, dw_w, t_dw);
  k_pw_wmma   <<<(B_*ZHW_)/(16*8), 256, 0, stream>>>(t_dw, whp, t_pw);      // 2048 blocks
  k_stats     <<<B_*C_, 256, 0, stream>>>(t_pw, st1);
  k_in1_apply <<<NVOX_/256, 256, 0, stream>>>(x, t_pw, st1, in1_g, in1_b, t_x2);
  k_ln        <<<T_/256, 256, 0, stream>>>(t_x2, ln_g, ln_b, t_seq);
  k_inproj_wmma<<<(T_/16)*4/8, 256, 0, stream>>>(t_seq, whi, t_xz);         // 8192 blocks
  k_scan      <<<NSEQ_, 128, 0, stream>>>(t_xz, c1w, c1b, xpw, dtw, dtb, alog, dskip, t_y);
  k_out_wmma  <<<(T_/16)/8, 256, 0, stream>>>(t_y, who, whj, projb, t_post);// 2048 blocks
  k_stats     <<<B_*C_, 256, 0, stream>>>(t_post, st2);
  k_post      <<<NVOX_/256, 256, 0, stream>>>(x, t_post, st2, postg, postb, out);
}
